// TransformerModel_37391985279608
// MI455X (gfx1250) — compile-verified, matrix operand order checked by reading, not run on hardware
//
#include <hip/hip_runtime.h>
#include <hip/hip_bf16.h>
#include <cstddef>
#include <cstdint>

// ---------------------------------------------------------------------------
// Types for CDNA5 WMMA (wave32): v_wmma_f32_16x16x32_bf16
// ---------------------------------------------------------------------------
typedef __attribute__((ext_vector_type(16))) __bf16       v16bf;
typedef __attribute__((ext_vector_type(8)))  float        v8f;
typedef __attribute__((ext_vector_type(8)))  unsigned int v8u;
typedef __attribute__((ext_vector_type(4)))  unsigned int u32x4;
typedef __attribute__((ext_vector_type(8)))  unsigned int u32x8;

#define HDc   512   // HEADS * HID
#define HEADSc  4
#define HIDc  128

// ---------------------------------------------------------------------------
// Helpers
// ---------------------------------------------------------------------------
__device__ __forceinline__ unsigned short f32_to_bf16_rne(float f) {
  unsigned int u = __float_as_uint(f);
  unsigned int r = u + 0x7FFFu + ((u >> 16) & 1u);
  return (unsigned short)(r >> 16);
}

__global__ void k_cvt_bf16(const float* __restrict__ in,
                           unsigned short* __restrict__ out, size_t n) {
  size_t i = (size_t)blockIdx.x * blockDim.x + threadIdx.x;
  if (i < n) out[i] = f32_to_bf16_rne(in[i]);
}

__global__ void k_fill_f32(float* __restrict__ p, float v, size_t n) {
  size_t i = (size_t)blockIdx.x * blockDim.x + threadIdx.x;
  if (i < n) p[i] = v;
}

// ---------------------------------------------------------------------------
// TDM: DMA one 64x32-half weight tile (rows = output cols, stride K halves)
// from global into LDS, letting the TDM apply the bank-conflict padding:
//   pad_interval code 3 = 16 dwords (64B = one 32-half row)
//   pad_amount   code 3 = 4 dwords  (16B) -> LDS row stride 40 halves.
// D# per CDNA5 ISA ch.8: group0 = {count, lds_addr, global_addr, type=2},
// group1 = {mask/data_size/pad, tensor_dim0/1, tile_dim0/1/2, dim0_stride}.
// Issued via inline asm (arity-portable across toolchains); completion is
// tracked with TENSORcnt (s_wait_tensorcnt).
// ---------------------------------------------------------------------------
__device__ __forceinline__ void tdm_load_tile(const unsigned short* gsrc,
                                              unsigned ldsOff, int K, int Ncols) {
  unsigned long long ga = (unsigned long long)(uintptr_t)gsrc;
  u32x4 g0;
  g0[0] = 1u;                                  // count=1, user descriptor
  g0[1] = ldsOff;                              // lds_addr (bytes)
  g0[2] = (unsigned)(ga & 0xFFFFFFFFu);        // global_addr[31:0]
  g0[3] = (unsigned)((ga >> 32) & 0x01FFFFFFu) // global_addr[56:32]
          | (2u << 30);                        // type = 2 (image)
  u32x8 g1;
  g1[0] = (1u << 16)                           // data_size = 2 bytes
        | (1u << 20)                           // pad_enable
        | (3u << 22)                           // pad_interval: 16 dwords
        | (3u << 25);                          // pad_amount:   4 dwords
  g1[1] = ((unsigned)K & 0xFFFFu) << 16;                         // tensor_dim0[15:0]
  g1[2] = (((unsigned)K >> 16) & 0xFFFFu)                        // tensor_dim0[31:16]
        | (((unsigned)Ncols & 0xFFFFu) << 16);                   // tensor_dim1[15:0]
  g1[3] = (((unsigned)Ncols >> 16) & 0xFFFFu) | (32u << 16);     // tile_dim0 = 32
  g1[4] = 64u;                                                   // tile_dim1 = 64
  g1[5] = (unsigned)K;                                           // dim0_stride[31:0]
  g1[6] = 0u;
  g1[7] = 0u;
  asm volatile("tensor_load_to_lds %0, %1" :: "s"(g0), "s"(g1) : "memory");
}

// ---------------------------------------------------------------------------
// bf16 WMMA GEMM:  C[M,Ncols] = A[M,K] * B[Ncols,K]^T + bias
// Block = 256 threads = 8 waves. Wave tile = 16x64 (4 wmma accumulators),
// block tile = 128x64. Weight tile double-buffered in LDS via TDM: wave 0
// issues tensor_load_to_lds for tile i+1, waits TENSORcnt<=1 (tile i done),
// barrier, all waves run 4 back-to-back WMMAs against LDS buffer i.
// A fragment layout (16-bit A 16x32, ISA 7.12.2):
//   lane 0-15 : M=lane,    halves 0-7 => K 0..7,  halves 8-15 => K 16..23
//   lane16-31 : M=lane-16, halves 0-7 => K 8..15, halves 8-15 => K 24..31
// B fragment mirrors it with N = lane&15 (weights are [Ncols,K] row-major).
// C/D layout: vgpr r, lane l -> row = r + 8*(l>=16), col = l&15.
// ---------------------------------------------------------------------------
__global__ __launch_bounds__(256) void k_gemm_bf16_wmma(
    const unsigned short* __restrict__ A,   // [M,K] bf16
    const unsigned short* __restrict__ B,   // [Ncols,K] bf16 (weight, row-major)
    const float* __restrict__ bias,         // [Ncols]
    float* __restrict__ C,                  // [M,Ncols] f32
    int M, int Ncols, int K)
{
  __shared__ alignas(16) unsigned short ldsB[2][64 * 40]; // [buf][col][40]

  const int lane    = threadIdx.x & 31;
  const int wave    = threadIdx.x >> 5;
  const int colBase = blockIdx.x * 64;
  const int rowBase = blockIdx.y * 128 + wave * 16;
  const bool mActive = rowBase < M;          // wave-uniform (M % 16 == 0)

  const int lr   = lane & 15;
  const int koff = (lane >> 4) * 8;

  const unsigned short* Btile = B + (size_t)colBase * K;
  const unsigned short* Arow  = A + (size_t)(rowBase + lr) * K + koff;
  const unsigned ldsOff0 = (unsigned)(uintptr_t)&ldsB[0][0];
  const unsigned ldsOff1 = (unsigned)(uintptr_t)&ldsB[1][0];

  v8f acc[4] = {v8f{}, v8f{}, v8f{}, v8f{}};

  const int T = K >> 5;                      // K / 32 tiles
  if (wave == 0) tdm_load_tile(Btile, ldsOff0, K, Ncols);   // prologue: tile 0

  for (int it = 0; it < T; ++it) {
    const int k0 = it << 5;
    if (wave == 0) {
      if (it + 1 < T) {                      // overlap DMA of next tile
        tdm_load_tile(Btile + (k0 + 32), ((it + 1) & 1) ? ldsOff1 : ldsOff0,
                      K, Ncols);
        __builtin_amdgcn_s_wait_tensorcnt(1);   // tile `it` complete
      } else {
        __builtin_amdgcn_s_wait_tensorcnt(0);
      }
    }
    __syncthreads();                         // publish LDS tile to all waves

    if (mActive) {
      const unsigned short* ldsTile = &ldsB[it & 1][0];
      uint4 a0 = *reinterpret_cast<const uint4*>(Arow + k0);
      uint4 a1 = *reinterpret_cast<const uint4*>(Arow + k0 + 16);
      if (k0 + 32 < K)                       // global_prefetch_b8 next A tile
        __builtin_prefetch((const void*)(Arow + k0 + 32), 0, 1);
      v8u araw;
      araw[0]=a0.x; araw[1]=a0.y; araw[2]=a0.z; araw[3]=a0.w;
      araw[4]=a1.x; araw[5]=a1.y; araw[6]=a1.z; araw[7]=a1.w;
      v16bf afrag = __builtin_bit_cast(v16bf, araw);

      // gather all four B fragments first: one ds wait, then 4 WMMAs b2b
      v16bf bfrag[4];
#pragma unroll
      for (int j = 0; j < 4; ++j) {
        const unsigned short* bp = ldsTile + (j * 16 + lr) * 40 + koff;
        uint4 b0 = *reinterpret_cast<const uint4*>(bp);
        uint4 b1 = *reinterpret_cast<const uint4*>(bp + 16);
        v8u braw;
        braw[0]=b0.x; braw[1]=b0.y; braw[2]=b0.z; braw[3]=b0.w;
        braw[4]=b1.x; braw[5]=b1.y; braw[6]=b1.z; braw[7]=b1.w;
        bfrag[j] = __builtin_bit_cast(v16bf, braw);
      }
#pragma unroll
      for (int j = 0; j < 4; ++j) {
        acc[j] = __builtin_amdgcn_wmma_f32_16x16x32_bf16(
            /*neg_a=*/false, afrag, /*neg_b=*/false, bfrag[j],
            /*c_mod=*/(short)0, acc[j], /*reuse_a=*/false, /*reuse_b=*/false);
      }
    }
    __syncthreads();                         // reads done before buf reuse
  }

  if (mActive) {
    const int rowOff = (lane >> 4) * 8;
#pragma unroll
    for (int j = 0; j < 4; ++j) {
      const int col = colBase + j * 16 + lr;
      const float bv = bias[col];
#pragma unroll
      for (int r = 0; r < 8; ++r) {
        const int row = rowBase + rowOff + r;
        C[(size_t)row * Ncols + col] = acc[j][r] + bv;
      }
    }
  }
}

// ---------------------------------------------------------------------------
// Edge kernels (wave-per-edge; one head per float4 sweep of 128 elements)
// ---------------------------------------------------------------------------
__global__ void k_edge_alpha(const float* __restrict__ q,  // [N,512]
                             const float* __restrict__ k,  // [N,512]
                             const int* __restrict__ src,
                             const int* __restrict__ dst,
                             float* __restrict__ alpha,    // [E,4]
                             int E)
{
  const int wid  = (int)(((size_t)blockIdx.x * blockDim.x + threadIdx.x) >> 5);
  const int lane = threadIdx.x & 31;
  if (wid >= E) return;                      // wave-uniform
  const int s = src[wid], d = dst[wid];
  const float4* q4 = reinterpret_cast<const float4*>(q + (size_t)d * HDc);
  const float4* k4 = reinterpret_cast<const float4*>(k + (size_t)s * HDc);
  float accs[HEADSc];
#pragma unroll
  for (int h = 0; h < HEADSc; ++h) {
    float4 a = q4[h * 32 + lane];
    float4 b = k4[h * 32 + lane];
    accs[h] = a.x * b.x + a.y * b.y + a.z * b.z + a.w * b.w;
  }
#pragma unroll
  for (int h = 0; h < HEADSc; ++h) {
    float v = accs[h];
#pragma unroll
    for (int off = 16; off > 0; off >>= 1) v += __shfl_xor(v, off, 32);
    if (lane == h) alpha[(size_t)wid * HEADSc + h] = v * 0.08838834764831845f; // 1/sqrt(128)
  }
}

// segment max via ordered-int atomic trick (init must be -inf)
__global__ void k_seg_max(const float* __restrict__ alpha,
                          const int* __restrict__ dst,
                          float* __restrict__ amax, int E4)
{
  int i = blockIdx.x * blockDim.x + threadIdx.x;
  if (i >= E4) return;
  const int e = i >> 2, h = i & 3;
  const float v = alpha[i];
  float* addr = amax + (size_t)dst[e] * HEADSc + h;
  if (v >= 0.0f) atomicMax((int*)addr, __float_as_int(v));
  else           atomicMin((unsigned int*)addr, (unsigned int)__float_as_int(v));
}

__global__ void k_fix_max(float* __restrict__ amax, size_t n) {
  size_t i = (size_t)blockIdx.x * blockDim.x + threadIdx.x;
  if (i < n && amax[i] == -__builtin_huge_valf()) amax[i] = 0.0f;
}

__global__ void k_exp_sum(const float* __restrict__ alpha,
                          const int* __restrict__ dst,
                          const float* __restrict__ amax,
                          float* __restrict__ ea,
                          float* __restrict__ asum, int E4)
{
  int i = blockIdx.x * blockDim.x + threadIdx.x;
  if (i >= E4) return;
  const int e = i >> 2, h = i & 3;
  const int d = dst[e];
  const float v = __expf(alpha[i] - amax[(size_t)d * HEADSc + h]);
  ea[i] = v;
  atomicAdd(asum + (size_t)d * HEADSc + h, v);
}

__global__ void k_aggregate(const float* __restrict__ ea,    // [E,4]
                            const float* __restrict__ asum,  // [N,4]
                            const float* __restrict__ vsrc,  // [N,512]
                            const int* __restrict__ src,
                            const int* __restrict__ dst,
                            float* __restrict__ agg,         // [N,512] (zeroed)
                            int E)
{
  const int wid  = (int)(((size_t)blockIdx.x * blockDim.x + threadIdx.x) >> 5);
  const int lane = threadIdx.x & 31;
  if (wid >= E) return;
  const int s = src[wid], d = dst[wid];
  const float4* v4 = reinterpret_cast<const float4*>(vsrc + (size_t)s * HDc);
  float* out = agg + (size_t)d * HDc;
#pragma unroll
  for (int h = 0; h < HEADSc; ++h) {
    const float coef = ea[(size_t)wid * HEADSc + h] /
                       (asum[(size_t)d * HEADSc + h] + 1e-16f);
    float4 vv = v4[h * 32 + lane];
    float* p = out + h * HIDc + lane * 4;
    atomicAdd(p + 0, coef * vv.x);
    atomicAdd(p + 1, coef * vv.y);
    atomicAdd(p + 2, coef * vv.z);
    atomicAdd(p + 3, coef * vv.w);
  }
}

// skip-connection + ReLU, emit bf16 activations for next WMMA GEMM
__global__ void k_finalize(const float* __restrict__ agg,
                           const float* __restrict__ sproj,
                           unsigned short* __restrict__ hb, size_t n)
{
  size_t i = (size_t)blockIdx.x * blockDim.x + threadIdx.x;
  if (i < n) hb[i] = f32_to_bf16_rne(fmaxf(agg[i] + sproj[i], 0.0f));
}

// ---------------------------------------------------------------------------
// Host orchestration
// ---------------------------------------------------------------------------
extern "C" void kernel_launch(void* const* d_in, const int* in_sizes, int n_in,
                              void* d_out, int out_size, void* d_ws, size_t ws_size,
                              hipStream_t stream) {
  (void)n_in; (void)out_size; (void)ws_size;

  const float* x  = (const float*)d_in[0];
  const int*   ei = (const int*)d_in[1];
  const size_t Nn = (size_t)in_sizes[0] / 384;   // 50000
  const size_t E  = (size_t)in_sizes[1] / 2;     // 800000
  const int* src = ei;
  const int* dst = ei + E;

  // weight/bias pointers: Wq0,bq0,Wk0,bk0,Wv0,bv0,Ws0,bs0, Wq1..bs1, Wo,bo
  const float* Wf[9]; const float* bf[9];
  for (int i = 0; i < 9; ++i) { Wf[i] = (const float*)d_in[2 + 2*i]; bf[i] = (const float*)d_in[3 + 2*i]; }
  const size_t Wsz[9] = { 512*384, 512*384, 512*384, 512*384,
                          512*512, 512*512, 512*512, 512*512,
                          384*512 };

  // workspace carve-out (256B aligned)
  char* wp = (char*)d_ws;
  auto alloc = [&](size_t bytes) -> void* {
    void* p = (void*)wp; wp += (bytes + 255) & ~(size_t)255; return p;
  };
  float* qb   = (float*)alloc(Nn * HDc * 4);
  float* kb   = (float*)alloc(Nn * HDc * 4);
  float* vb   = (float*)alloc(Nn * HDc * 4);
  float* spb  = (float*)alloc(Nn * HDc * 4);
  float* agg  = (float*)alloc(Nn * HDc * 4);
  float* alp  = (float*)alloc(E * HEADSc * 4);
  float* ea   = (float*)alloc(E * HEADSc * 4);
  float* amax = (float*)alloc(Nn * HEADSc * 4);
  float* asum = (float*)alloc(Nn * HEADSc * 4);
  unsigned short* xb = (unsigned short*)alloc(Nn * 384 * 2);
  unsigned short* hb = (unsigned short*)alloc(Nn * HDc * 2);
  unsigned short* Wb[9];
  for (int i = 0; i < 9; ++i) Wb[i] = (unsigned short*)alloc(Wsz[i] * 2);

  auto nblk = [](size_t n) { return (unsigned)((n + 255) / 256); };

  // --- convert activations & weights to bf16 ---
  k_cvt_bf16<<<nblk(Nn * 384), 256, 0, stream>>>(x, xb, Nn * 384);
  for (int i = 0; i < 9; ++i)
    k_cvt_bf16<<<nblk(Wsz[i]), 256, 0, stream>>>(Wf[i], Wb[i], Wsz[i]);

  auto gemm = [&](const unsigned short* A, const unsigned short* Bw,
                  const float* bias, float* C, size_t M, int Ncols, int K) {
    dim3 g((unsigned)(Ncols / 64), (unsigned)((M + 127) / 128));
    k_gemm_bf16_wmma<<<g, 256, 0, stream>>>(A, Bw, bias, C, (int)M, Ncols, K);
  };

  auto attention = [&](const float* q, const float* k, const float* v,
                       const float* sp, unsigned short* hout) {
    k_fill_f32<<<nblk(Nn * HEADSc), 256, 0, stream>>>(amax, -__builtin_huge_valf(), Nn * HEADSc);
    k_fill_f32<<<nblk(Nn * HEADSc), 256, 0, stream>>>(asum, 0.0f, Nn * HEADSc);
    k_fill_f32<<<nblk(Nn * HDc),    256, 0, stream>>>(agg,  0.0f, Nn * HDc);
    k_edge_alpha<<<nblk(E * 32), 256, 0, stream>>>(q, k, src, dst, alp, (int)E);
    k_seg_max  <<<nblk(E * HEADSc), 256, 0, stream>>>(alp, dst, amax, (int)(E * HEADSc));
    k_fix_max  <<<nblk(Nn * HEADSc), 256, 0, stream>>>(amax, Nn * HEADSc);
    k_exp_sum  <<<nblk(E * HEADSc), 256, 0, stream>>>(alp, dst, amax, ea, asum, (int)(E * HEADSc));
    k_aggregate<<<nblk(E * 32), 256, 0, stream>>>(ea, asum, v, src, dst, agg, (int)E);
    k_finalize <<<nblk(Nn * HDc), 256, 0, stream>>>(agg, sp, hout, Nn * HDc);
  };

  // ---- layer 0: projections (K=384) + attention ----
  gemm(xb, Wb[0], bf[0], qb,  Nn, 512, 384);
  gemm(xb, Wb[1], bf[1], kb,  Nn, 512, 384);
  gemm(xb, Wb[2], bf[2], vb,  Nn, 512, 384);
  gemm(xb, Wb[3], bf[3], spb, Nn, 512, 384);
  attention(qb, kb, vb, spb, hb);

  // ---- layer 1: projections (K=512) + attention ----
  gemm(hb, Wb[4], bf[4], qb,  Nn, 512, 512);
  gemm(hb, Wb[5], bf[5], kb,  Nn, 512, 512);
  gemm(hb, Wb[6], bf[6], vb,  Nn, 512, 512);
  gemm(hb, Wb[7], bf[7], spb, Nn, 512, 512);
  attention(qb, kb, vb, spb, hb);   // hb safely overwritten after its last read

  // ---- output projection: [N,384] = h @ Wo^T + bo ----
  gemm(hb, Wb[8], bf[8], (float*)d_out, Nn, 384, 512);
}
